// GRUDecoder_31568009625740
// MI455X (gfx1250) — compile-verified, matrix operand order checked by reading, not run on hardware
//
#include <hip/hip_runtime.h>
#include <hip/hip_bf16.h>
#include <cstddef>

// ---------------------------------------------------------------------------
// Problem constants (from the reference)
// ---------------------------------------------------------------------------
constexpr int cB  = 64;        // batch
constexpr int cT  = 2048;      // time steps
constexpr int cIN = 512;       // input features
constexpr int cH  = 128;       // hidden
constexpr int cG3 = 3 * cH;    // 384 = gates (r,z,n)
constexpr int cM  = cB * cT;   // 131072 rows of the hoisted projection GEMMs

typedef __attribute__((ext_vector_type(16))) __bf16 v16bf;
typedef __attribute__((ext_vector_type(8)))  __bf16 v8bf;
typedef __attribute__((ext_vector_type(8)))  float  v8f;

// ---------------------------------------------------------------------------
// WMMA fragment loader (16x32 bf16 A-tile or 32x16 bf16 B-tile, row stride in
// elements).  Per ISA 7.12.2: lanes 0-15 hold K={0..7,16..23}, lanes 16-31
// hold K={8..15,24..31} for row/col (lane&15).  16B-aligned vector loads.
// ---------------------------------------------------------------------------
__device__ __forceinline__ v16bf load_frag(const __bf16* tile_base, int stride, int lane) {
    int row   = lane & 15;
    int khalf = (lane >> 4) & 1;
    const __bf16* p = tile_base + row * stride + khalf * 8;
    v8bf lo = *(const v8bf*)p;
    v8bf hi = *(const v8bf*)(p + 16);
    return __builtin_shufflevector(lo, hi, 0,1,2,3,4,5,6,7,8,9,10,11,12,13,14,15);
}

__device__ __forceinline__ float sigmoidf_fast(float x) { return 1.f / (1.f + __expf(-x)); }
__device__ __forceinline__ float mishf(float x) { return x * tanhf(log1pf(__expf(x))); }

// ---------------------------------------------------------------------------
// fp32 -> bf16 conversion kernels
// ---------------------------------------------------------------------------
__global__ void cvt_w_kernel(const float* __restrict__ src, __bf16* __restrict__ dst, int n) {
    for (int i = blockIdx.x * blockDim.x + threadIdx.x; i < n; i += gridDim.x * blockDim.x)
        dst[i] = (__bf16)src[i];
}

// x [B,T,IN] fp32  ->  xbf [T,B,IN] bf16  (so projection rows are m = t*B + b)
__global__ void cvt_x_kernel(const float* __restrict__ x, __bf16* __restrict__ xbf) {
    const size_t N = (size_t)cB * cT * cIN;
    for (size_t idx = (size_t)blockIdx.x * blockDim.x + threadIdx.x; idx < N;
         idx += (size_t)gridDim.x * blockDim.x) {
        size_t b   = idx / ((size_t)cT * cIN);
        size_t rem = idx % ((size_t)cT * cIN);
        size_t t   = rem / cIN;
        size_t i   = rem % cIN;
        xbf[(t * cB + b) * cIN + i] = (__bf16)x[idx];
    }
}

// ---------------------------------------------------------------------------
// Hoisted input-projection GEMM:  out[m,n] = sum_k A[m,k]*W[n,k] + bias[n]
// A is bf16 [M,K0] (optionally concat with A1 [M,K1] along k), W bf16 [N,Ktot].
// One wave computes a 16x64 strip (4 N-tiles share one A fragment).
// Grid: exactly (M/16)*(384/64)/8 blocks of 256 threads (8 waves).
// ---------------------------------------------------------------------------
__global__ void proj_kernel(const __bf16* __restrict__ A0, const __bf16* __restrict__ A1,
                            int K0, int K1, int Ktot,
                            const __bf16* __restrict__ Wb, const float* __restrict__ bias,
                            float* __restrict__ out) {
    const int lane = threadIdx.x & 31;
    const int wave = threadIdx.x >> 5;
    const int wid  = blockIdx.x * 8 + wave;
    const int mt   = wid / 6;        // 8192 M-tiles
    const int ns   = wid % 6;        // 6 strips of 64 over N=384

    v8f acc[4] = {{}, {}, {}, {}};
    for (int kb = 0; kb < Ktot; kb += 32) {
        const __bf16* pa;
        int strA;
        if (kb < K0) { pa = A0 + (size_t)mt * 16 * K0 + kb;        strA = K0; }
        else         { pa = A1 + (size_t)mt * 16 * K1 + (kb - K0); strA = K1; }
        v16bf a = load_frag(pa, strA, lane);
#pragma unroll
        for (int nt = 0; nt < 4; ++nt) {
            const __bf16* pb = Wb + (size_t)(ns * 64 + nt * 16) * Ktot + kb;
            v16bf bfrag = load_frag(pb, Ktot, lane);
            acc[nt] = __builtin_amdgcn_wmma_f32_16x16x32_bf16(
                false, a, false, bfrag, (short)0, acc[nt], false, false);
        }
    }
    const int hi8 = ((lane >> 4) & 1) * 8;   // C/D layout: lanes>=16 hold M=8..15
#pragma unroll
    for (int nt = 0; nt < 4; ++nt) {
        int n = ns * 64 + nt * 16 + (lane & 15);
        float bn = bias[n];
#pragma unroll
        for (int r = 0; r < 8; ++r)
            out[(size_t)(mt * 16 + r + hi8) * cG3 + n] = acc[nt][r] + bn;
    }
}

// ---------------------------------------------------------------------------
// Recurrent scan.  One workgroup (8 waves) per direction; h (fp32 + bf16) and
// the full Whh (bf16) live in LDS for all 2048 steps.  Per step:
//   gh[64,384] = h_bf16 @ Whh^T
// Tile map: 2 waves per M-tile (wave>>1), 12 N-tiles each; the 4 A fragments
// (hidden state) are hoisted into registers once per step and reused for all
// 12 N-tiles -> 12x less A-side LDS traffic on the latency-critical path.
// ---------------------------------------------------------------------------
struct ScanArgs {
    const float*  xp[2];       // [M,384] precomputed x-projections (+bih)
    const __bf16* whh[2];      // [384,128] bf16
    const float*  bhh[2];      // [384]
    __bf16*       out_seq[2];  // [M,128] bf16 or nullptr
    float*        h_out[2];    // [64,128] fp32 or nullptr
    int           rev[2];
};

constexpr int sHP = 136;  // bf16 row stride for h    (64  x 136)
constexpr int sWP = 136;  // bf16 row stride for Whh  (384 x 136)
constexpr int sGP = 396;  // f32  row stride for gh   (64  x 396) - conflict-free stores
constexpr int sFP = 132;  // f32  row stride for h
constexpr int LDS_WHH = 384 * sWP * 2;              // 104448
constexpr int LDS_HBF = 64 * sHP * 2;               // 17408
constexpr int LDS_GH  = 64 * sGP * 4;               // 101376
constexpr int LDS_HF  = 64 * sFP * 4;               // 33792
constexpr int LDS_BH  = 384 * 4;                    // 1536
constexpr int SCAN_LDS_BYTES = LDS_WHH + LDS_HBF + LDS_GH + LDS_HF + LDS_BH;  // 258560

__global__ void gru_scan_kernel(ScanArgs args) {
    extern __shared__ __align__(16) char smem[];
    __bf16* whl = (__bf16*)smem;
    __bf16* hbf = (__bf16*)(smem + LDS_WHH);
    float*  gh  = (float*)(smem + LDS_WHH + LDS_HBF);
    float*  hf  = (float*)(smem + LDS_WHH + LDS_HBF + LDS_GH);
    float*  bh  = (float*)(smem + LDS_WHH + LDS_HBF + LDS_GH + LDS_HF);

    const int d = blockIdx.x;
    const float*  xp    = args.xp[d];
    const __bf16* whh_g = args.whh[d];
    const float*  bhh_g = args.bhh[d];
    __bf16*       oseq  = args.out_seq[d];
    float*        hout  = args.h_out[d];
    const int     rev   = args.rev[d];

    const int tid  = threadIdx.x;
    const int lane = tid & 31;
    const int wave = tid >> 5;
    const int mt   = wave >> 1;         // 4 M-tiles, 2 waves each
    const int nt0  = (wave & 1) * 12;   // each wave owns 12 N-tiles

    // init: h = 0, stage Whh + bhh into LDS (reused for all 2048 steps)
    for (int i = tid; i < 64 * sHP;  i += 256) hbf[i] = (__bf16)0.f;
    for (int i = tid; i < 64 * sFP;  i += 256) hf[i]  = 0.f;
    for (int i = tid; i < 384 * 128; i += 256) whl[(i >> 7) * sWP + (i & 127)] = whh_g[i];
    for (int i = tid; i < 384;       i += 256) bh[i]  = bhh_g[i];
    __syncthreads();

    const int hi8 = ((lane >> 4) & 1) * 8;
    for (int it = 0; it < cT; ++it) {
        const int t = rev ? (cT - 1 - it) : it;
        const float* xrow = xp + (size_t)t * (cB * cG3);

        // prefetch this step's gate pre-activations behind the WMMA phase
        {
            const char* px = (const char*)xrow;
            for (int o = tid * 128; o < cB * cG3 * 4; o += 256 * 128)
                __builtin_prefetch(px + o, 0, 0);
        }

        // hoist the hidden-state A fragments (16x128) into registers
        v16bf afr[4];
#pragma unroll
        for (int kc = 0; kc < 4; ++kc)
            afr[kc] = load_frag(hbf + mt * 16 * sHP + kc * 32, sHP, lane);

        // gh = h @ Whh^T over this wave's 12 N-tiles
#pragma unroll 4
        for (int nn = 0; nn < 12; ++nn) {
            const int nt = nt0 + nn;
            v8f acc = {};
#pragma unroll
            for (int kc = 0; kc < 4; ++kc) {
                v16bf b = load_frag(whl + nt * 16 * sWP + kc * 32, sWP, lane);
                acc = __builtin_amdgcn_wmma_f32_16x16x32_bf16(
                    false, afr[kc], false, b, (short)0, acc, false, false);
            }
            const int col = nt * 16 + (lane & 15);
#pragma unroll
            for (int r = 0; r < 8; ++r)
                gh[(mt * 16 + r + hi8) * sGP + col] = acc[r];
        }
        __syncthreads();

        // fp32 GRU cell (torch gate order r,z,n)
        for (int e = tid; e < cB * cH; e += 256) {
            const int b = e >> 7, j = e & 127;
            const float hr = gh[b * sGP + j]        + bh[j];
            const float hz = gh[b * sGP + 128 + j]  + bh[128 + j];
            const float hn = gh[b * sGP + 256 + j]  + bh[256 + j];
            const float xr = xrow[(size_t)b * cG3 + j];
            const float xz = xrow[(size_t)b * cG3 + 128 + j];
            const float xn = xrow[(size_t)b * cG3 + 256 + j];
            const float r = sigmoidf_fast(xr + hr);
            const float z = sigmoidf_fast(xz + hz);
            const float n = tanhf(xn + r * hn);
            const float hnew = (1.f - z) * n + z * hf[b * sFP + j];
            hf[b * sFP + j]  = hnew;
            hbf[b * sHP + j] = (__bf16)hnew;
            if (oseq) oseq[((size_t)t * cB + b) * cH + j] = (__bf16)hnew;
        }
        __syncthreads();
    }

    if (hout)
        for (int e = tid; e < cB * cH; e += 256)
            hout[e] = hf[(e >> 7) * sFP + (e & 127)];
}

// ---------------------------------------------------------------------------
// Head: mish -> LN(g1,be1) -> Linear(W1,b1) -> mish -> LN(g2,be2) -> Linear(W2,b2)
// All fp32, single block, [64,128] in LDS.
// ---------------------------------------------------------------------------
__global__ void head_kernel(const float* __restrict__ hin,
                            const float* __restrict__ g1, const float* __restrict__ be1,
                            const float* __restrict__ W1, const float* __restrict__ b1,
                            const float* __restrict__ g2, const float* __restrict__ be2,
                            const float* __restrict__ W2, const float* __restrict__ b2,
                            float* __restrict__ out) {
    __shared__ float a[64][132];
    const int tid = threadIdx.x;

    for (int e = tid; e < 64 * 128; e += 256) a[e >> 7][e & 127] = mishf(hin[e]);
    __syncthreads();

    if (tid < 64) {   // LayerNorm 1
        float m = 0.f;
        for (int k = 0; k < 128; ++k) m += a[tid][k];
        m *= (1.f / 128.f);
        float v = 0.f;
        for (int k = 0; k < 128; ++k) { float dlt = a[tid][k] - m; v += dlt * dlt; }
        v *= (1.f / 128.f);
        float inv = rsqrtf(v + 1e-5f);
        for (int k = 0; k < 128; ++k) a[tid][k] = (a[tid][k] - m) * inv * g1[k] + be1[k];
    }
    __syncthreads();

    float creg[32];   // Linear1 + mish (read everything before overwriting LDS)
#pragma unroll 4
    for (int s = 0; s < 32; ++s) {
        const int e = tid + s * 256, r = e >> 7, j = e & 127;
        float acc = b1[j];
        for (int k = 0; k < 128; ++k) acc += a[r][k] * W1[j * 128 + k];
        creg[s] = mishf(acc);
    }
    __syncthreads();
    for (int s = 0; s < 32; ++s) { const int e = tid + s * 256; a[e >> 7][e & 127] = creg[s]; }
    __syncthreads();

    if (tid < 64) {   // LayerNorm 2
        float m = 0.f;
        for (int k = 0; k < 128; ++k) m += a[tid][k];
        m *= (1.f / 128.f);
        float v = 0.f;
        for (int k = 0; k < 128; ++k) { float dlt = a[tid][k] - m; v += dlt * dlt; }
        v *= (1.f / 128.f);
        float inv = rsqrtf(v + 1e-5f);
        for (int k = 0; k < 128; ++k) a[tid][k] = (a[tid][k] - m) * inv * g2[k] + be2[k];
    }
    __syncthreads();

    if (tid < 64) {   // Final Linear(H,1)
        float acc = b2[0];
        for (int k = 0; k < 128; ++k) acc += a[tid][k] * W2[k];
        out[tid] = acc;
    }
}

// ---------------------------------------------------------------------------
// Host orchestration
// ---------------------------------------------------------------------------
extern "C" void kernel_launch(void* const* d_in, const int* in_sizes, int n_in,
                              void* d_out, int out_size, void* d_ws, size_t ws_size,
                              hipStream_t stream) {
    (void)in_sizes; (void)n_in; (void)out_size; (void)ws_size;
    const float* x     = (const float*)d_in[0];
    const float* Wih0f = (const float*)d_in[1];
    const float* Whh0f = (const float*)d_in[2];
    const float* bih0f = (const float*)d_in[3];
    const float* bhh0f = (const float*)d_in[4];
    const float* Wih0b = (const float*)d_in[5];
    const float* Whh0b = (const float*)d_in[6];
    const float* bih0b = (const float*)d_in[7];
    const float* bhh0b = (const float*)d_in[8];
    // d_in[9..12] = layer-1 forward GRU: dead code in the reference (hf unused)
    const float* Wih1b = (const float*)d_in[13];
    const float* Whh1b = (const float*)d_in[14];
    const float* bih1b = (const float*)d_in[15];
    const float* bhh1b = (const float*)d_in[16];
    const float* g1  = (const float*)d_in[17];
    const float* be1 = (const float*)d_in[18];
    const float* W1  = (const float*)d_in[19];
    const float* b1  = (const float*)d_in[20];
    const float* g2  = (const float*)d_in[21];
    const float* be2 = (const float*)d_in[22];
    const float* W2  = (const float*)d_in[23];
    const float* b2  = (const float*)d_in[24];

    char* ws = (char*)d_ws;
    size_t off = 0;
    auto carve = [&](size_t bytes) { char* p = ws + off; off += (bytes + 255) & ~(size_t)255; return p; };

    __bf16* xbf   = (__bf16*)carve((size_t)cM * cIN * 2);   // 128 MB (fits in 192 MB L2)
    float*  xp0f  = (float*) carve((size_t)cM * cG3 * 4);   // 192 MB
    float*  xp0b  = (float*) carve((size_t)cM * cG3 * 4);   // 192 MB
    __bf16* wih0f_b = (__bf16*)carve((size_t)cG3 * cIN * 2);
    __bf16* wih0b_b = (__bf16*)carve((size_t)cG3 * cIN * 2);
    __bf16* wih1b_b = (__bf16*)carve((size_t)cG3 * (2 * cH) * 2);
    __bf16* whh0f_b = (__bf16*)carve((size_t)cG3 * cH * 2);
    __bf16* whh0b_b = (__bf16*)carve((size_t)cG3 * cH * 2);
    __bf16* whh1b_b = (__bf16*)carve((size_t)cG3 * cH * 2);
    float*  hfinal  = (float*) carve((size_t)cB * cH * 4);
    // aliases: xbf is dead after the layer-0 projections -> reuse for of/ob;
    // xp0f is dead after scan0 -> reuse for xp1b.
    __bf16* ofb  = (__bf16*)xbf;
    __bf16* obb  = (__bf16*)((char*)xbf + (size_t)cM * cH * 2);
    float*  xp1b = xp0f;

    hipFuncSetAttribute((const void*)gru_scan_kernel,
                        hipFuncAttributeMaxDynamicSharedMemorySize, SCAN_LDS_BYTES);

    // 1) weight + input conversions to bf16
    cvt_w_kernel<<<768, 256, 0, stream>>>(Wih0f, wih0f_b, cG3 * cIN);
    cvt_w_kernel<<<768, 256, 0, stream>>>(Wih0b, wih0b_b, cG3 * cIN);
    cvt_w_kernel<<<384, 256, 0, stream>>>(Wih1b, wih1b_b, cG3 * 2 * cH);
    cvt_w_kernel<<<192, 256, 0, stream>>>(Whh0f, whh0f_b, cG3 * cH);
    cvt_w_kernel<<<192, 256, 0, stream>>>(Whh0b, whh0b_b, cG3 * cH);
    cvt_w_kernel<<<192, 256, 0, stream>>>(Whh1b, whh1b_b, cG3 * cH);
    cvt_x_kernel<<<4096, 256, 0, stream>>>(x, xbf);

    // 2) hoisted layer-0 input projections (big WMMA GEMMs, + bih fold-in)
    const int projBlocks = (cM / 16) * 6 / 8;   // 6144
    proj_kernel<<<projBlocks, 256, 0, stream>>>(xbf, nullptr, cIN, 0, cIN, wih0f_b, bih0f, xp0f);
    proj_kernel<<<projBlocks, 256, 0, stream>>>(xbf, nullptr, cIN, 0, cIN, wih0b_b, bih0b, xp0b);

    // 3) layer-0 recurrent scans, fwd + bwd concurrently (one WGP each)
    ScanArgs s0;
    s0.xp[0] = xp0f;  s0.whh[0] = whh0f_b;  s0.bhh[0] = bhh0f;
    s0.out_seq[0] = ofb;  s0.h_out[0] = nullptr;  s0.rev[0] = 0;
    s0.xp[1] = xp0b;  s0.whh[1] = whh0b_b;  s0.bhh[1] = bhh0b;
    s0.out_seq[1] = obb;  s0.h_out[1] = nullptr;  s0.rev[1] = 1;
    gru_scan_kernel<<<2, 256, SCAN_LDS_BYTES, stream>>>(s0);

    // 4) layer-1 backward input projection over concat(of, ob)
    proj_kernel<<<projBlocks, 256, 0, stream>>>(ofb, obb, cH, cH, 2 * cH, wih1b_b, bih1b, xp1b);

    // 5) layer-1 backward scan; only the final hidden state is needed
    ScanArgs s1;
    s1.xp[0] = xp1b;  s1.whh[0] = whh1b_b;  s1.bhh[0] = bhh1b;
    s1.out_seq[0] = nullptr;  s1.h_out[0] = hfinal;  s1.rev[0] = 1;
    s1.xp[1] = xp1b;  s1.whh[1] = whh1b_b;  s1.bhh[1] = bhh1b;
    s1.out_seq[1] = nullptr;  s1.h_out[1] = nullptr; s1.rev[1] = 1;
    gru_scan_kernel<<<1, 256, SCAN_LDS_BYTES, stream>>>(s1);

    // 6) head
    head_kernel<<<1, 256, 0, stream>>>(hfinal, g1, be1, W1, b1, g2, be2, W2, b2,
                                       (float*)d_out);
}
